// HZTrajGRUCell_17188459118835
// MI455X (gfx1250) — matrix operand order repaired and checked by reading, not compiled, not run on hardware
//
#include <hip/hip_runtime.h>
#include <hip/hip_bf16.h>
#include <math.h>

// Problem constants (from reference): S=8, B=8, C_IN=16, H=W=96, NF=64, L=5
#define WDIM   96
#define HW     9216
#define NFEAT  64
#define SLEN   8
#define BATCH  8
#define LINKS  5

typedef __attribute__((ext_vector_type(16))) __bf16 v16bf;
typedef __attribute__((ext_vector_type(8)))  float  v8f;

union Reg8 { unsigned u[8]; v16bf v; };

static __device__ __forceinline__ unsigned f32_to_bf16_rne(float x) {
  union { float f; unsigned u; } c; c.f = x;
  unsigned u = c.u;
  unsigned r = u + 0x7FFFu + ((u >> 16) & 1u);
  return r >> 16;
}
static __device__ __forceinline__ unsigned pack_bf16(float a, float b) {
  return f32_to_bf16_rne(a) | (f32_to_bf16_rne(b) << 16);
}

// ---------------------------------------------------------------------------
// Pack OIHW f32 weights into WMMA B-operand bf16 pair layout:
// Bp[kpair][n] (dword) = { bf16(K=2*kpair) | bf16(K=2*kpair+1) << 16 }
// K index = ci*KH*KW + ky*KW + kx (matches the A-side im2col gather order).
// ---------------------------------------------------------------------------
__global__ __launch_bounds__(256)
void pack_weights(const float* __restrict__ w, unsigned* __restrict__ bp,
                  int C_out, int K_real, int Npad, int total)
{
  int id = blockIdx.x * 256 + threadIdx.x;
  if (id >= total) return;
  int kp = id / Npad, n = id % Npad;
  float a = 0.f, b = 0.f;
  int k0 = 2 * kp;
  if (n < C_out) {
    if (k0 < K_real)     a = w[(size_t)n * K_real + k0];
    if (k0 + 1 < K_real) b = w[(size_t)n * K_real + k0 + 1];
  }
  bp[id] = pack_bf16(a, b);
}

__global__ __launch_bounds__(256)
void zero_f32(float* __restrict__ p, size_t n)
{
  size_t i = (size_t)blockIdx.x * 256 + threadIdx.x;
  if (i < n) p[i] = 0.f;
}

// ---------------------------------------------------------------------------
// Generic implicit-GEMM conv via v_wmma_f32_16x16x32_bf16.
// Template KH (compile-time kernel size -> constant-divisor im2col math) and
// NT (N-tiles per wave: wave computes a 16 x 16*NT tile, reusing the gathered
// A operand NT times so the matrix pipe amortizes the scattered A fetch).
// 8 waves/block: wave w -> m-tile (blockIdx.x*8+w), shared 16*NT n-slab per
// block (blockIdx.y), image blockIdx.z. B chunks staged through LDS.
// ---------------------------------------------------------------------------
template <int KH, int NT>
__global__ __launch_bounds__(256)
void conv_wmma(const float* __restrict__ in, const unsigned* __restrict__ Bp,
               const float* __restrict__ bias, float* __restrict__ out,
               int C_in, int C_out, int pad, int Kpad, int Npad,
               int in_img_stride, int out_img_stride, int accum, int act)
{
  constexpr int KHW = KH * KH;
  constexpr int NW  = 16 * NT;            // n-slab width per block
  __shared__ unsigned lb[16 * NW];        // 16 kpairs x NW dwords

  const int lane = threadIdx.x & 31;
  const int wave = threadIdx.x >> 5;
  const int img  = blockIdx.z;
  const int m_base = (blockIdx.x * 8 + wave) * 16;
  const int n_base = blockIdx.y * NW;
  const int K_real = C_in * KHW;
  const int nk     = Kpad >> 5;

  const float* __restrict__ inp = in + (size_t)img * in_img_stride;

  const int h   = lane >> 4;              // half select (K offset)
  const int l16 = lane & 15;
  const int mA  = m_base + l16;           // A-matrix row gathered by this lane
  const int py  = mA / WDIM, px = mA % WDIM;

  v8f acc[NT];
#pragma unroll
  for (int nt = 0; nt < NT; ++nt) acc[nt] = (v8f){};

  for (int kc = 0; kc < nk; ++kc) {
    // Stage 32-K weight chunk (16 kpairs x NW n dwords) into LDS.
#pragma unroll
    for (int i = 0; i < NT; ++i) {
      int flat = i * 256 + threadIdx.x;
      int kpl = flat / NW, nl = flat % NW;
      lb[flat] = Bp[(size_t)(kc * 16 + kpl) * Npad + (n_base + nl)];
      if (i == 0 && kc + 1 < nk)
        __builtin_prefetch(&Bp[(size_t)((kc + 1) * 16 + kpl) * Npad + (n_base + nl)], 0, 1);
    }
    __syncthreads();

    // A operand: lane(m=lane%16, h=lane/16); vgpr j<4 -> K=8h+2j, j>=4 -> K=16+8h+2(j-4)
    Reg8 a;
#pragma unroll
    for (int j = 0; j < 8; ++j) {
      int kpair = (j < 4) ? (4 * h + j) : (8 + 4 * h + (j - 4));
      int k0 = kc * 32 + kpair * 2;
      float f0 = 0.f, f1 = 0.f;
      if (k0 < K_real) {
        int ci = k0 / KHW, r = k0 - ci * KHW;   // constant-divisor (templated)
        int dy = r / KH,   dx = r - dy * KH;
        int yy = py - pad + dy, xx = px - pad + dx;
        if (yy >= 0 && yy < WDIM && xx >= 0 && xx < WDIM)
          f0 = inp[(size_t)ci * HW + yy * WDIM + xx];
      }
      int k1 = k0 + 1;
      if (k1 < K_real) {
        int ci = k1 / KHW, r = k1 - ci * KHW;
        int dy = r / KH,   dx = r - dy * KH;
        int yy = py - pad + dy, xx = px - pad + dx;
        if (yy >= 0 && yy < WDIM && xx >= 0 && xx < WDIM)
          f1 = inp[(size_t)ci * HW + yy * WDIM + xx];
      }
      a.u[j] = pack_bf16(f0, f1);
    }

    // NT matrix ops reuse the gathered A operand.
    // B operand: lane(n=lane%16, h=lane/16); vgpr j<4 holds kpair 4h+j, j>=4 holds kpair 8+4h+(j-4)
#pragma unroll
    for (int nt = 0; nt < NT; ++nt) {
      Reg8 b;
#pragma unroll
      for (int j = 0; j < 8; ++j)
        b.u[j] = lb[((j < 4) ? (4 * h + j) : (8 + 4 * h + (j - 4))) * NW + nt * 16 + l16];
      acc[nt] = __builtin_amdgcn_wmma_f32_16x16x32_bf16(false, a.v, false, b.v,
                                                        (short)0, acc[nt], false, false);
    }
    __syncthreads();
  }

  // Epilogue. D layout: lane(n=lane%16, h=lane/16); vgpr j -> m = 8h + j.
#pragma unroll
  for (int nt = 0; nt < NT; ++nt) {
    int n = n_base + nt * 16 + l16;
    if (n < C_out) {
      float bv = bias[n];
      float* __restrict__ op = out + (size_t)img * out_img_stride + (size_t)n * HW;
#pragma unroll
      for (int j = 0; j < 8; ++j) {
        int m = m_base + 8 * h + j;
        float v = acc[nt][j] + bv;
        if (accum) v += op[m];
        if (act)   v = tanhf(v);
        op[m] = v;
      }
    }
  }
}

// ---------------------------------------------------------------------------
// Bilinear warp of h_prev by -flow (grid_sample, zero padding).
// One thread per (b, link, pixel); loops 64 channels reusing corner weights.
// px = (gx - flow_x) * W/(W-1) - 0.5  (matches reference normalization chain)
// ---------------------------------------------------------------------------
__global__ __launch_bounds__(256)
void warp_bilinear(const float* __restrict__ hprev, const float* __restrict__ flows,
                   float* __restrict__ warped)
{
  int id = blockIdx.x * 256 + threadIdx.x;     // over B*L*HW
  int p  = id % HW;
  int bl = id / HW;
  int l  = bl % LINKS, b = bl / LINKS;

  const float* fl = flows + ((size_t)b * (2 * LINKS) + l * 2) * HW;
  float fx = fl[p], fy = fl[HW + p];
  float gx = (float)(p % WDIM), gy = (float)(p / WDIM);
  const float SCL = (float)WDIM / (float)(WDIM - 1);
  float pxf = (gx - fx) * SCL - 0.5f;
  float pyf = (gy - fy) * SCL - 0.5f;

  float x0f = floorf(pxf), y0f = floorf(pyf);
  int x0 = (int)x0f, y0 = (int)y0f, x1 = x0 + 1, y1 = y0 + 1;
  float wx1 = pxf - x0f, wx0 = 1.f - wx1;
  float wy1 = pyf - y0f, wy0 = 1.f - wy1;

  float v00 = (x0 >= 0 && x0 < WDIM && y0 >= 0 && y0 < WDIM) ? 1.f : 0.f;
  float v01 = (x1 >= 0 && x1 < WDIM && y0 >= 0 && y0 < WDIM) ? 1.f : 0.f;
  float v10 = (x0 >= 0 && x0 < WDIM && y1 >= 0 && y1 < WDIM) ? 1.f : 0.f;
  float v11 = (x1 >= 0 && x1 < WDIM && y1 >= 0 && y1 < WDIM) ? 1.f : 0.f;
  int x0c = min(max(x0, 0), WDIM - 1), x1c = min(max(x1, 0), WDIM - 1);
  int y0c = min(max(y0, 0), WDIM - 1), y1c = min(max(y1, 0), WDIM - 1);
  float w00 = wy0 * wx0 * v00, w01 = wy0 * wx1 * v01;
  float w10 = wy1 * wx0 * v10, w11 = wy1 * wx1 * v11;
  int o00 = y0c * WDIM + x0c, o01 = y0c * WDIM + x1c;
  int o10 = y1c * WDIM + x0c, o11 = y1c * WDIM + x1c;

  const float* hb = hprev + (size_t)b * NFEAT * HW;
  float* wb = warped + ((size_t)b * (LINKS * NFEAT) + l * NFEAT) * HW + p;
#pragma unroll 4
  for (int c = 0; c < NFEAT; ++c) {
    const float* hc = hb + (size_t)c * HW;
    wb[(size_t)c * HW] = w00 * hc[o00] + w01 * hc[o01]
                       + w10 * hc[o10] + w11 * hc[o11];
  }
}

// ---------------------------------------------------------------------------
// GRU gating: r,u = sigmoid; m = tanh(i2h_m + r*h2h_m); h = u*h_prev+(1-u)*m
// ---------------------------------------------------------------------------
__global__ __launch_bounds__(256)
void gru_gate(const float* __restrict__ i2h, const float* __restrict__ h2h,
              const float* __restrict__ hprev, float* __restrict__ hnew)
{
  int id = blockIdx.x * 256 + threadIdx.x;     // over B*NF*HW
  int p  = id % HW;
  int bc = id / HW;
  int c  = bc % NFEAT, b = bc / NFEAT;

  size_t base = (size_t)b * (3 * NFEAT) * HW;
  size_t rC = base + (size_t)c * HW + p;
  size_t uC = base + (size_t)(c + NFEAT) * HW + p;
  size_t mC = base + (size_t)(c + 2 * NFEAT) * HW + p;

  float r = 1.f / (1.f + expf(-(i2h[rC] + h2h[rC])));
  float u = 1.f / (1.f + expf(-(i2h[uC] + h2h[uC])));
  float m = tanhf(i2h[mC] + r * h2h[mC]);

  size_t hi = ((size_t)b * NFEAT + c) * HW + p;
  hnew[hi] = u * hprev[hi] + (1.f - u) * m;
}

// ---------------------------------------------------------------------------
extern "C" void kernel_launch(void* const* d_in, const int* in_sizes, int n_in,
                              void* d_out, int out_size, void* d_ws, size_t ws_size,
                              hipStream_t stream)
{
  (void)in_sizes; (void)n_in; (void)out_size; (void)ws_size;

  const float* inputs = (const float*)d_in[0];
  const float* i2h_w  = (const float*)d_in[1];
  const float* i2h_b  = (const float*)d_in[2];
  const float* i2f_w  = (const float*)d_in[3];
  const float* i2f_b  = (const float*)d_in[4];
  const float* h2f_w  = (const float*)d_in[5];
  const float* h2f_b  = (const float*)d_in[6];
  const float* fl_w   = (const float*)d_in[7];
  const float* fl_b   = (const float*)d_in[8];
  const float* ret_w  = (const float*)d_in[9];
  const float* ret_b  = (const float*)d_in[10];
  // d_in[11] = seq_len (== SLEN for this reference)

  // Workspace carving (256B aligned)
  char* ws = (char*)d_ws;
  size_t off = 0;
  auto carve = [&](size_t bytes) -> char* {
    char* p = ws + off;
    off += (bytes + 255) & ~(size_t)255;
    return p;
  };
  unsigned* wb_i2h = (unsigned*)carve((size_t)80  * 192 * 4);  // Kpad=160
  unsigned* wb_i2f = (unsigned*)carve((size_t)208 * 32  * 4);  // Kpad=416
  unsigned* wb_h2f = (unsigned*)carve((size_t)800 * 32  * 4);  // Kpad=1600
  unsigned* wb_fl  = (unsigned*)carve((size_t)400 * 16  * 4);  // Kpad=800
  unsigned* wb_ret = (unsigned*)carve((size_t)160 * 192 * 4);  // Kpad=320
  float* h0    = (float*)carve((size_t)BATCH * NFEAT * HW * 4);
  float* i2h_t = (float*)carve((size_t)BATCH * 192   * HW * 4);
  float* fbuf  = (float*)carve((size_t)BATCH * 32    * HW * 4);
  float* flbuf = (float*)carve((size_t)BATCH * 10    * HW * 4);
  float* wbuf  = (float*)carve((size_t)BATCH * 320   * HW * 4);
  float* h2h   = (float*)carve((size_t)BATCH * 192   * HW * 4);

  // Pack all weights to bf16 B-operand layout
  auto pack = [&](const float* w, unsigned* bp, int cout, int kreal, int kpad, int npad) {
    int total = (kpad / 2) * npad;
    pack_weights<<<dim3((total + 255) / 256), dim3(256), 0, stream>>>(w, bp, cout, kreal, npad, total);
  };
  pack(i2h_w, wb_i2h, 192, 144,  160,  192);
  pack(i2f_w, wb_i2f, 32,  400,  416,  32);
  pack(h2f_w, wb_h2f, 32,  1600, 1600, 32);
  pack(fl_w,  wb_fl,  10,  800,  800,  16);
  pack(ret_w, wb_ret, 192, 320,  320,  192);

  const size_t hstate = (size_t)BATCH * NFEAT * HW;
  zero_f32<<<dim3((unsigned)((hstate + 255) / 256)), dim3(256), 0, stream>>>(h0, hstate);

  float* outs = (float*)d_out;

  for (int t = 0; t < SLEN; ++t) {
    const float* x_t = inputs + (size_t)t * BATCH * 16 * HW;
    const float* hp  = (t == 0) ? h0 : (outs + (size_t)(t - 1) * hstate);

    // i2h: 16 -> 192, 3x3 pad1   (3 n-slabs of 64)
    conv_wmma<3, 4><<<dim3(72, 3, 8), dim3(256), 0, stream>>>(
        x_t, wb_i2h, i2h_b, i2h_t, 16, 192, 1, 160, 192, 16 * HW, 192 * HW, 0, 0);
    // i2f: 16 -> 32, 5x5 pad2    (1 n-slab of 32)
    conv_wmma<5, 2><<<dim3(72, 1, 8), dim3(256), 0, stream>>>(
        x_t, wb_i2f, i2f_b, fbuf, 16, 32, 2, 416, 32, 16 * HW, 32 * HW, 0, 0);
    // h2f: 64 -> 32, 5x5 pad2, accumulate into fbuf + tanh => f
    conv_wmma<5, 2><<<dim3(72, 1, 8), dim3(256), 0, stream>>>(
        hp, wb_h2f, h2f_b, fbuf, 64, 32, 2, 1600, 32, 64 * HW, 32 * HW, 1, 1);
    // flows: 32 -> 10, 5x5 pad2  (1 n-slab of 16)
    conv_wmma<5, 1><<<dim3(72, 1, 8), dim3(256), 0, stream>>>(
        fbuf, wb_fl, fl_b, flbuf, 32, 10, 2, 800, 16, 32 * HW, 10 * HW, 0, 0);
    // bilinear warp of h_prev along 5 links
    warp_bilinear<<<dim3(1440), dim3(256), 0, stream>>>(hp, flbuf, wbuf);
    // ret: 320 -> 192, 1x1       (3 n-slabs of 64)
    conv_wmma<1, 4><<<dim3(72, 3, 8), dim3(256), 0, stream>>>(
        wbuf, wb_ret, ret_b, h2h, 320, 192, 0, 320, 192, 320 * HW, 192 * HW, 0, 0);
    // GRU gates -> outs[t] (doubles as next h_prev)
    gru_gate<<<dim3(18432), dim3(256), 0, stream>>>(i2h_t, h2h, hp, outs + (size_t)t * hstate);
  }

  // h_last = outs[S-1]
  hipMemcpyAsync(outs + (size_t)SLEN * hstate, outs + (size_t)(SLEN - 1) * hstate,
                 hstate * sizeof(float), hipMemcpyDeviceToDevice, stream);
}
